// OccupancyNet_64776696758882
// MI455X (gfx1250) — matrix operand order repaired and checked by reading, not compiled, not run on hardware
//
#include <hip/hip_runtime.h>
#include <math.h>

typedef __attribute__((ext_vector_type(16))) _Float16 v16h;
typedef __attribute__((ext_vector_type(8)))  float    v8f;

#define GD 128
#define GH 128
#define GW 32
#define GC 64

// ---------------------------------------------------------------------------
// Prep: convert w1 (64x64) and w2 (64x64) fp32 row-major [K][N] into f16
// WMMA B-fragment order for V_WMMA_F32_16X16X32_F16:
//   fragment f = nt*2 + kc  (nt = N-tile 0..3, kc = K-chunk 0..1)
//   half index within fragment = lane*16 + j
//   K = (lane>=16 ? 8 : 0) + (j&7) + (j>=8 ? 16 : 0) + 32*kc ,  N = nt*16 + (lane&15)
// Layout in ws: [layer(2)][frag(8)][lane(32)][j(16)] halfs  -> 16 KB total.
// ---------------------------------------------------------------------------
__global__ void occ_prep(const float* __restrict__ w1,
                         const float* __restrict__ w2,
                         _Float16* __restrict__ wfrag) {
  for (int idx = threadIdx.x; idx < 2 * 8 * 512; idx += blockDim.x) {
    int layer  = idx >> 12;          // /4096
    int rem    = idx & 4095;
    int f      = rem >> 9;           // /512
    int within = rem & 511;
    int lane   = within >> 4;
    int j      = within & 15;
    int nt = f >> 1, kc = f & 1;
    int K = ((lane >> 4) ? 8 : 0) + (j & 7) + ((j >> 3) ? 16 : 0) + 32 * kc;
    int N = nt * 16 + (lane & 15);
    const float* w = layer ? w2 : w1;
    wfrag[idx] = (_Float16)w[K * 64 + N];
  }
}

// ---------------------------------------------------------------------------
// Main: one wave handles 16 queries. 256 threads = 8 waves / block.
// ---------------------------------------------------------------------------
__global__ __launch_bounds__(256) void occ_main(
    const float* __restrict__ feats, const unsigned char* __restrict__ valid,
    const float* __restrict__ queries,
    const float* __restrict__ b1, const float* __restrict__ b2,
    const float* __restrict__ w3, const float* __restrict__ b3,
    const _Float16* __restrict__ wfrag,
    float* __restrict__ out_val, float* __restrict__ out_ok,
    float* __restrict__ out_logit, int Q, long nWaves) {
  __shared__ _Float16 lds[8][16 * 64];

  const int lane = threadIdx.x & 31;
  const int wid  = threadIdx.x >> 5;
  const long wglobal = (long)blockIdx.x * 8 + wid;
  if (wglobal >= nWaves) return;                  // wave-uniform

  const int m  = lane & 15;                       // query row handled
  const int hi = lane >> 4;                       // lane half (0/1)
  const int kb = hi * 8;                          // channel group offset

  const long gq = wglobal * 16 + m;               // global query id
  const int  b  = (int)(gq / Q);
  const long qi = gq % Q;

  // --- query position -> grid coords -------------------------------------
  const float* qp = queries + ((long)b * Q + qi) * 3;
  const float fx = qp[0] * 2.0f;                  // / CELL_SIZE(0.5)
  const float fy = qp[1] * 2.0f;
  const float fz = qp[2] * 2.0f;
  const float f0 = floorf(fx), f1 = floorf(fy), f2 = floorf(fz);
  const int   i0 = (int)f0,   i1 = (int)f1,   i2 = (int)f2;
  const float r0 = fx - f0,   r1 = fy - f1,   r2 = fz - f2;

  // --- trilinear gather directly into A-fragment element order -----------
  // acc[a], a = kc*16 + j  <->  channel = kb + 16*(a/8) + (a%8)
  float acc[32];
#pragma unroll
  for (int i = 0; i < 32; ++i) acc[i] = 0.0f;
  bool ok = true;

  const float*         fb = feats + (long)b * GD * GH * GW * GC;
  const unsigned char* vb = valid + (long)b * GD * GH * GW;

#pragma unroll
  for (int dx = 0; dx < 2; ++dx)
#pragma unroll
    for (int dy = 0; dy < 2; ++dy)
#pragma unroll
      for (int dz = 0; dz < 2; ++dz) {
        const int cx = i0 + dx, cy = i1 + dy, cz = i2 + dz;
        const bool inb = (cx >= 0) & (cx < GD) & (cy >= 0) & (cy < GH) &
                         (cz >= 0) & (cz < GW);
        const int px = cx < 0 ? 0 : (cx > GD - 1 ? GD - 1 : cx);
        const int py = cy < 0 ? 0 : (cy > GH - 1 ? GH - 1 : cy);
        const int pz = cz < 0 ? 0 : (cz > GW - 1 ? GW - 1 : cz);
        const float wgt = (dx ? r0 : 1.0f - r0) * (dy ? r1 : 1.0f - r1) *
                          (dz ? r2 : 1.0f - r2);
        const long cell = ((long)px * GH + py) * GW + pz;
        const float* fp = fb + cell * GC + kb;
        ok = ok && inb && (vb[cell] != 0);
#pragma unroll
        for (int t = 0; t < 4; ++t) {             // 32B contiguous chunks
          const float4 lo = *(const float4*)(fp + 16 * t);
          const float4 hv = *(const float4*)(fp + 16 * t + 4);
          acc[t * 8 + 0] += wgt * lo.x;
          acc[t * 8 + 1] += wgt * lo.y;
          acc[t * 8 + 2] += wgt * lo.z;
          acc[t * 8 + 3] += wgt * lo.w;
          acc[t * 8 + 4] += wgt * hv.x;
          acc[t * 8 + 5] += wgt * hv.y;
          acc[t * 8 + 6] += wgt * hv.z;
          acc[t * 8 + 7] += wgt * hv.w;
        }
      }

  v16h a0, a1;
#pragma unroll
  for (int j = 0; j < 16; ++j) {
    a0[j] = (_Float16)acc[j];
    a1[j] = (_Float16)acc[16 + j];
  }

  // --- layer 1: h1 = relu(feats @ w1 + b1)  via WMMA ----------------------
  const v16h* B1 = (const v16h*)wfrag;            // 8 fragments
  const v16h* B2 = B1 + 256;                      // +4096 halfs
  v8f ct[4];
#pragma unroll
  for (int nt = 0; nt < 4; ++nt) {
    const v16h bA = B1[(nt * 2 + 0) * 32 + lane];
    const v16h bB = B1[(nt * 2 + 1) * 32 + lane];
    v8f c = {};
    c = __builtin_amdgcn_wmma_f32_16x16x32_f16(false, a0, false, bA,
                                               (short)0, c, false, false);
    c = __builtin_amdgcn_wmma_f32_16x16x32_f16(false, a1, false, bB,
                                               (short)0, c, false, false);
    const float bias = b1[nt * 16 + m];
#pragma unroll
    for (int r = 0; r < 8; ++r) {
      const float v = c[r] + bias;
      c[r] = v > 0.0f ? v : 0.0f;
    }
    ct[nt] = c;
  }

  // --- reshape h1 (C-fragment) -> A-fragment via LDS ----------------------
  {
    _Float16* Lw = lds[wid];
#pragma unroll
    for (int nt = 0; nt < 4; ++nt)
#pragma unroll
      for (int r = 0; r < 8; ++r)
        Lw[(r + hi * 8) * 64 + nt * 16 + m] = (_Float16)ct[nt][r];
  }
  __syncthreads();

  v16h a20, a21;
  {
    const _Float16* Lr = lds[wid] + m * 64 + kb;
#pragma unroll
    for (int j = 0; j < 8; ++j) {
      a20[j]     = Lr[j];        // t=0: ch kb+0..7
      a20[8 + j] = Lr[16 + j];   // t=1: ch kb+16..23
      a21[j]     = Lr[32 + j];   // t=2
      a21[8 + j] = Lr[48 + j];   // t=3
    }
  }

  // --- layer 2: h2 = relu(h1 @ w2 + b2) -----------------------------------
  v8f c2[4];
#pragma unroll
  for (int nt = 0; nt < 4; ++nt) {
    const v16h bA = B2[(nt * 2 + 0) * 32 + lane];
    const v16h bB = B2[(nt * 2 + 1) * 32 + lane];
    v8f c = {};
    c = __builtin_amdgcn_wmma_f32_16x16x32_f16(false, a20, false, bA,
                                               (short)0, c, false, false);
    c = __builtin_amdgcn_wmma_f32_16x16x32_f16(false, a21, false, bB,
                                               (short)0, c, false, false);
    const float bias = b2[nt * 16 + m];
#pragma unroll
    for (int r = 0; r < 8; ++r) {
      const float v = c[r] + bias;
      c[r] = v > 0.0f ? v : 0.0f;
    }
    c2[nt] = c;
  }

  // --- layer 3: logits = h2 @ w3 + b3 (dot + cross-lane reduce) -----------
  float w3v[4];
#pragma unroll
  for (int nt = 0; nt < 4; ++nt) w3v[nt] = w3[nt * 16 + m];

  float part[8];
#pragma unroll
  for (int r = 0; r < 8; ++r)
    part[r] = c2[0][r] * w3v[0] + c2[1][r] * w3v[1] +
              c2[2][r] * w3v[2] + c2[3][r] * w3v[3];

#pragma unroll
  for (int mask = 1; mask < 16; mask <<= 1)
#pragma unroll
    for (int r = 0; r < 8; ++r)
      part[r] += __shfl_xor(part[r], mask);

  // ok: lane L (L<16) owns query m
  if (hi == 0) out_ok[wglobal * 16 + m] = ok ? 1.0f : 0.0f;

  // rows: lane (m<8) in half `hi` writes query m + hi*8 with part[m]
  if (m < 8) {
    const long g2 = wglobal * 16 + (m + hi * 8);
    const float lg = part[m] + b3[0];
    out_logit[g2] = lg;
    out_val[g2]   = 1.0f / (1.0f + expf(-lg));
  }
}

extern "C" void kernel_launch(void* const* d_in, const int* in_sizes, int n_in,
                              void* d_out, int out_size, void* d_ws,
                              size_t ws_size, hipStream_t stream) {
  const float*         feats   = (const float*)d_in[0];
  const unsigned char* valid   = (const unsigned char*)d_in[1];
  const float*         queries = (const float*)d_in[2];
  const float*         w1      = (const float*)d_in[3];
  const float*         b1      = (const float*)d_in[4];
  const float*         w2      = (const float*)d_in[5];
  const float*         b2      = (const float*)d_in[6];
  const float*         w3      = (const float*)d_in[7];
  const float*         b3      = (const float*)d_in[8];

  const int BQ = out_size / 3;     // values ++ ok ++ logits
  const int Q  = BQ / 4;           // B = 4
  float* out_val   = (float*)d_out;
  float* out_ok    = out_val + BQ;
  float* out_logit = out_ok + BQ;

  _Float16* wfrag = (_Float16*)d_ws;   // 16 KB used
  occ_prep<<<1, 256, 0, stream>>>(w1, w2, wfrag);

  const long nWaves = (long)BQ / 16;
  const int  blocks = (int)((nWaves + 7) / 8);
  occ_main<<<blocks, 256, 0, stream>>>(feats, valid, queries, b1, b2, w3, b3,
                                       wfrag, out_val, out_ok, out_logit, Q,
                                       nWaves);
}
